// Hypernetwork_7060926234846
// MI455X (gfx1250) — matrix operand order repaired and checked by reading, not compiled, not run on hardware
//
#include <hip/hip_runtime.h>
#include <hip/hip_bf16.h>

typedef __attribute__((ext_vector_type(16))) _Float16 v16h;
typedef __attribute__((ext_vector_type(8)))  float    v8f;

#define INP  4096
#define MSK  64
#define HIDD 64
#define OUTD 10
#define WVEC 4810
#define NT4  301   // ceil(4810/16)
#define KT4  4     // 128/32

// ---------------- prep: tiled transpose W1 (64 x 4096) -> W1T (4096 x 64) ----------------
__global__ __launch_bounds__(256) void k_transpose_w1(const float* __restrict__ W1,
                                                      float* __restrict__ W1T) {
  __shared__ float tile[32][33];
  int bx = blockIdx.x & 127;   // 4096/32 col tiles
  int by = blockIdx.x >> 7;    // 64/32  row tiles
  int x = bx * 32 + threadIdx.x;            // col in W1
#pragma unroll
  for (int k = 0; k < 32; k += 8) {
    int y = by * 32 + threadIdx.y + k;      // row in W1
    tile[threadIdx.y + k][threadIdx.x] = W1[y * 4096 + x];
  }
  __syncthreads();
  int xo = by * 32 + threadIdx.x;           // col of W1T (width 64)
#pragma unroll
  for (int k = 0; k < 32; k += 8) {
    int yo = bx * 32 + threadIdx.y + k;     // row of W1T
    W1T[yo * 64 + xo] = tile[threadIdx.x][threadIdx.y + k];
  }
}

// ---------------- prep: pack W (N x K, f32) into f16 WMMA B-layout tiles ----------------
// packed index: (((nt*kTiles + kt)*32 + lane)*16 + e), element = W[n*K + k],
// n = nt*16 + (lane&15), k = kt*32 + (e<8?0:16) + (lane>>4)*8 + (e&7) ; zero-pad n >= N.
__global__ __launch_bounds__(256) void k_pack_b(const float* __restrict__ W,
                                                _Float16* __restrict__ pk,
                                                int N, int K, int nTiles, int kTiles) {
  int t = blockIdx.x * blockDim.x + threadIdx.x;     // one thread per (tile, lane)
  int total = nTiles * kTiles * 32;
  if (t >= total) return;
  int lane = t & 31;
  int tile = t >> 5;
  int kt = tile % kTiles;
  int nt = tile / kTiles;
  int n  = nt * 16 + (lane & 15);
  int hi = lane >> 4;
  v16h v;
#pragma unroll
  for (int e = 0; e < 16; ++e) {
    int k = kt * 32 + ((e < 8) ? 0 : 16) + hi * 8 + (e & 7);
    v[e] = (n < N) ? (_Float16)W[(size_t)n * K + k] : (_Float16)0.f;
  }
  ((v16h*)pk)[t] = v;
}

// ---------------- device helpers ----------------
// A-matrix (16x32 f16) per-lane layout: m = lane&15, hi = lane>>4,
// element e -> k_local = (e<8?0:16) + hi*8 + (e&7)
__device__ __forceinline__ v16h loadA(const _Float16* base, int ld, int m, int hi, int kbase) {
  v16h a;
#pragma unroll
  for (int e = 0; e < 16; ++e) {
    int kl = ((e < 8) ? 0 : 16) + hi * 8 + (e & 7);
    a[e] = base[m * ld + kbase + kl];
  }
  return a;
}

__device__ __forceinline__ v8f wmma_f16(v16h a, v16h b, v8f c) {
  return __builtin_amdgcn_wmma_f32_16x16x32_f16(false, a, false, b, (short)0, c, false, false);
}

__device__ __forceinline__ float hred16(float v) {   // sum across 16-lane half-wave
  v += __shfl_xor(v, 1, 16);
  v += __shfl_xor(v, 2, 16);
  v += __shfl_xor(v, 4, 16);
  v += __shfl_xor(v, 8, 16);
  return v;
}

// ---------------- fused main kernel: 32 rows per workgroup, 16 waves ----------------
// B tiles are loaded once and reused across the two 16-row M-tiles (2x B reuse).
__global__ __launch_bounds__(512) void k_hyper_main(
    const float* __restrict__ data, const int* __restrict__ midx,
    const float* __restrict__ w1t,  const float* __restrict__ b1,
    const _Float16* __restrict__ pk2, const float* __restrict__ b2,
    const _Float16* __restrict__ pk3, const float* __restrict__ b3,
    const _Float16* __restrict__ pk4, const float* __restrict__ b4,
    float* __restrict__ out) {
  __shared__ int      s_idx[32][64];
  __shared__ float    s_xm[32][64];
  __shared__ _Float16 s_h1[32][64];
  __shared__ _Float16 s_h2[32][256];
  __shared__ _Float16 s_h3[32][128];
  __shared__ float    s_hid[32][64];
  __shared__ float    s_res[32][16];

  const int t    = threadIdx.x;
  const int lane = t & 31;
  const int wv   = t >> 5;          // 16 waves
  const int nn   = lane & 15;
  const int hi   = lane >> 4;
  const int row0 = blockIdx.x * 32;

  // init accumulators, load indices
  for (int i = t; i < 2048; i += 512) (&s_hid[0][0])[i] = 0.f;
  (&s_res[0][0])[t] = 0.f;
  for (int i = t; i < 2048; i += 512)
    s_idx[i >> 6][i & 63] = midx[(size_t)(row0 + (i >> 6)) * 64 + (i & 63)];
  __syncthreads();

  // stage 0: gather xm; h1 = relu(one_hot_sum(W1T) + b1) with dedup of sorted indices
  {
    const int j = t & 63;
    for (int r = t >> 6; r < 32; r += 8) {
      int idx = s_idx[r][j];
      s_xm[r][j] = data[(size_t)(row0 + r) * INP + idx];
      float acc = b1[j];
      int prev = -1;
      for (int m = 0; m < 64; ++m) {
        int id = s_idx[r][m];
        if (id != prev) acc += w1t[id * 64 + j];
        prev = id;
      }
      s_h1[r][j] = (_Float16)fmaxf(acc, 0.f);
    }
  }
  __syncthreads();

  // stage 1: h2 = relu(h1 @ W2^T + b2)  (M32, K64, N256: wave wv -> n-tile wv, both m-tiles)
  {
    int nt = wv;                                   // 16 n-tiles, 16 waves
    v16h a[2][2];
#pragma unroll
    for (int mt = 0; mt < 2; ++mt) {
      a[mt][0] = loadA(&s_h1[0][0], 64, mt * 16 + nn, hi, 0);
      a[mt][1] = loadA(&s_h1[0][0], 64, mt * 16 + nn, hi, 32);
    }
    const v16h* B2 = (const v16h*)pk2;
    v16h b0 = B2[(nt * 2 + 0) * 32 + lane];
    v16h b1v = B2[(nt * 2 + 1) * 32 + lane];
    float bias = b2[nt * 16 + nn];
#pragma unroll
    for (int mt = 0; mt < 2; ++mt) {
      v8f c;
#pragma unroll
      for (int r = 0; r < 8; ++r) c[r] = bias;
      c = wmma_f16(a[mt][0], b0, c);
      c = wmma_f16(a[mt][1], b1v, c);
#pragma unroll
      for (int r = 0; r < 8; ++r)
        s_h2[mt * 16 + r + hi * 8][nt * 16 + nn] = (_Float16)fmaxf(c[r], 0.f);
    }
  }
  __syncthreads();

  // stage 2: h3 = relu(h2 @ W3^T + b3)  (M32, K256, N128: wave -> (n-tile, m-tile))
  {
    int nt = wv & 7;                               // 8 n-tiles x 2 m-tiles = 16 waves
    int mt = wv >> 3;
    float bias = b3[nt * 16 + nn];
    v8f c;
#pragma unroll
    for (int r = 0; r < 8; ++r) c[r] = bias;
    const v16h* B3 = (const v16h*)pk3;
#pragma unroll
    for (int kt = 0; kt < 8; ++kt) {
      v16h a = loadA(&s_h2[0][0], 256, mt * 16 + nn, hi, kt * 32);
      c = wmma_f16(a, B3[(nt * 8 + kt) * 32 + lane], c);
    }
#pragma unroll
    for (int r = 0; r < 8; ++r)
      s_h3[mt * 16 + r + hi * 8][nt * 16 + nn] = (_Float16)fmaxf(c[r], 0.f);
  }
  __syncthreads();

  // stage 3: fused  w = h3 @ W4^T + b4  consumed tile-by-tile; B loaded once per n-tile,
  // reused for both m-tiles (loop-invariant A held in registers: 2 x 4 x v16h)
  v16h a4[2][KT4];
#pragma unroll
  for (int mt = 0; mt < 2; ++mt)
#pragma unroll
    for (int kt = 0; kt < KT4; ++kt)
      a4[mt][kt] = loadA(&s_h3[0][0], 128, mt * 16 + nn, hi, kt * 32);
  const v16h* B4 = (const v16h*)pk4;

  // pass A: inp_w (nt 0..255) and inp_b (nt 256..259) -> s_hid accumulation
  for (int nt = wv; nt < 260; nt += 16) {
    float bias = b4[nt * 16 + nn];
    v16h bt[KT4];
#pragma unroll
    for (int kt = 0; kt < KT4; ++kt) bt[kt] = B4[(nt * KT4 + kt) * 32 + lane];
#pragma unroll
    for (int mt = 0; mt < 2; ++mt) {
      v8f c;
#pragma unroll
      for (int r = 0; r < 8; ++r) c[r] = bias;
#pragma unroll
      for (int kt = 0; kt < KT4; ++kt) c = wmma_f16(a4[mt][kt], bt[kt], c);
      if (nt < 256) {                       // inp_w[h][m] * xm[m]
        int h  = nt >> 2;
        int m0 = (nt & 3) * 16;
#pragma unroll
        for (int r = 0; r < 8; ++r) {
          int rr = mt * 16 + r + hi * 8;
          float v = hred16(c[r] * s_xm[rr][m0 + nn]);
          if (nn == 0) atomicAdd(&s_hid[rr][h], v);
        }
      } else {                              // inp_b
        int h0 = (nt - 256) * 16;
#pragma unroll
        for (int r = 0; r < 8; ++r)
          atomicAdd(&s_hid[mt * 16 + r + hi * 8][h0 + nn], c[r]);
      }
    }
  }
  __syncthreads();
  for (int i = t; i < 2048; i += 512) {
    float* p = &s_hid[0][0];
    p[i] = fmaxf(p[i], 0.f);              // hid = relu(acc)
  }
  __syncthreads();

  // pass B: out_w (nt 260..299) and out_b (nt 300) -> s_res accumulation
  for (int nt = 260 + wv; nt < 301; nt += 16) {
    int cg0 = nt * 16;
    float bias = (cg0 + nn < WVEC) ? b4[cg0 + nn] : 0.f;
    v16h bt[KT4];
#pragma unroll
    for (int kt = 0; kt < KT4; ++kt) bt[kt] = B4[(nt * KT4 + kt) * 32 + lane];
#pragma unroll
    for (int mt = 0; mt < 2; ++mt) {
      v8f c;
#pragma unroll
      for (int r = 0; r < 8; ++r) c[r] = bias;
#pragma unroll
      for (int kt = 0; kt < KT4; ++kt) c = wmma_f16(a4[mt][kt], bt[kt], c);
      if (nt < 300) {                       // out_w[o][h] * hid[h]
        int o   = (nt - 260) >> 2;
        int hh0 = ((nt - 260) & 3) * 16;
#pragma unroll
        for (int r = 0; r < 8; ++r) {
          int rr = mt * 16 + r + hi * 8;
          float v = hred16(c[r] * s_hid[rr][hh0 + nn]);
          if (nn == 0) atomicAdd(&s_res[rr][o], v);
        }
      } else {                              // out_b
        if (nn < 10) {
#pragma unroll
          for (int r = 0; r < 8; ++r)
            atomicAdd(&s_res[mt * 16 + r + hi * 8][nn], c[r]);
        }
      }
    }
  }
  __syncthreads();
  for (int i = t; i < 320; i += 512)
    out[(size_t)(row0 + i / 10) * OUTD + (i % 10)] = s_res[i / 10][i % 10];
}

// ---------------- host launcher ----------------
extern "C" void kernel_launch(void* const* d_in, const int* in_sizes, int n_in,
                              void* d_out, int out_size, void* d_ws, size_t ws_size,
                              hipStream_t stream) {
  const float* data = (const float*)d_in[0];
  const int*   midx = (const int*)d_in[1];
  const float* W1   = (const float*)d_in[2];
  const float* b1   = (const float*)d_in[3];
  const float* W2   = (const float*)d_in[4];
  const float* b2   = (const float*)d_in[5];
  const float* W3   = (const float*)d_in[6];
  const float* b3   = (const float*)d_in[7];
  const float* W4   = (const float*)d_in[8];
  const float* b4   = (const float*)d_in[9];
  float* out = (float*)d_out;

  const int B = in_sizes[0] / INP;   // 16384

  char* ws = (char*)d_ws;
  float*    w1t = (float*)ws;                                       // 4096*64*4 = 1 MB
  _Float16* pk2 = (_Float16*)(ws + 1048576);                        // 16*2*512 f16 = 32 KB
  _Float16* pk3 = (_Float16*)(ws + 1048576 + 32768);                // 8*8*512 f16  = 64 KB
  _Float16* pk4 = (_Float16*)(ws + 1048576 + 32768 + 65536);        // 301*4*512 f16 ~ 1.2 MB

  dim3 tb(32, 8);
  k_transpose_w1<<<256, tb, 0, stream>>>(W1, w1t);
  k_pack_b<<<(16 * 2 * 32 + 255) / 256, 256, 0, stream>>>(W2, pk2, 256, 64, 16, 2);
  k_pack_b<<<(8 * 8 * 32 + 255) / 256, 256, 0, stream>>>(W3, pk3, 128, 256, 8, 8);
  k_pack_b<<<(NT4 * KT4 * 32 + 255) / 256, 256, 0, stream>>>(W4, pk4, WVEC, 128, NT4, KT4);

  k_hyper_main<<<B / 32, 512, 0, stream>>>(data, midx, w1t, b1,
                                           pk2, b2, pk3, b3, pk4, b4, out);
}
